// DimeNet_80917183857002
// MI455X (gfx1250) — matrix operand order repaired and checked by reading, not compiled, not run on hardware
//
#include <hip/hip_runtime.h>

// ---------------------------------------------------------------------------
// DimeNet-style fused GNN for gfx1250 (MI455X): bf16 WMMA GEMMs with fp32
// accumulate, fused gather (concat) -> GEMM -> ReLU -> scatter-atomic.
// ---------------------------------------------------------------------------

typedef __attribute__((ext_vector_type(16))) __bf16 v16bf;
typedef __attribute__((ext_vector_type(8)))  __bf16 v8bf;
typedef __attribute__((ext_vector_type(8)))  float  v8f;

__device__ __forceinline__ void atomic_add_f32(float* p, float v) {
  __hip_atomic_fetch_add(p, v, __ATOMIC_RELAXED, __HIP_MEMORY_SCOPE_AGENT);
}

// ---------------------------------------------------------------------------
// Pack an fp32 weight matrix [K, 256] into WMMA-B fragment-major bf16.
// Fragment layout (16x16x32 bf16 B, 32x16): for k-tile kt, col-tile nt:
//   lanes 0-15 : halves 0..7 = K = kt*32 + 0..7 , 8..15 = +16..23, col = lane
//   lanes 16-31: halves 0..7 = K = kt*32 + 8..15, 8..15 = +24..31, col = lane-16
// packed[(((kt*16+nt)*32 + lane)*16 + h]  (K zero-padded to KT*32)
// ---------------------------------------------------------------------------
__global__ void pack_weight_bf16(const float* __restrict__ W,
                                 __bf16* __restrict__ out, int K, int KT) {
  int idx = blockIdx.x * blockDim.x + threadIdx.x;
  int total = KT * 16 * 512;
  if (idx >= total) return;
  int h    = idx & 15;
  int lane = (idx >> 4) & 31;
  int tile = idx >> 9;           // kt*16 + nt
  int nt = tile & 15;
  int kt = tile >> 4;
  int col = lane & 15;
  int k = kt * 32 + ((lane < 16) ? 0 : 8) + ((h < 8) ? h : (8 + h));
  int n = nt * 16 + col;
  float v = (k < K) ? W[(long)k * 256 + n] : 0.0f;
  out[idx] = (__bf16)v;
}

// ---------------------------------------------------------------------------
// Fused gather -> bf16 WMMA GEMM (M=16 rows/block, N=256, K=KT*32) -> epilogue.
// MODE 0 (triplet): A = [h[k_idx[t]] | rbf[j_idx[t]] | cbf[t] | pad]
//                   out: ReLU, atomicAdd into agg_e[j_idx[t]]
// MODE 1 (edge):    A = [h[src[e]] | agg_e[dst[e]]]
//                   out: ReLU, atomicAdd into node_aggr[dst[e]]
// MODE 2 (node):    A = [h[i] | node_aggr[i]]
//                   out: + bias, direct store to h_next[i]
// 8 waves/block; each wave owns two 16-col tiles.
// ---------------------------------------------------------------------------
template <int MODE>
__global__ __launch_bounds__(256)
void fused_gather_gemm_scatter(
    const float* __restrict__ hsrc, int hld, int in_c,
    const float* __restrict__ rbf,  const float* __restrict__ cbf,
    const int*   __restrict__ kidx, const int*   __restrict__ jidx,
    const int*   __restrict__ srcp, const int*   __restrict__ dstp,
    const float* __restrict__ aux,
    const __bf16* __restrict__ wp, int KT,
    const float* __restrict__ bias,
    float* __restrict__ outbuf, int nrows)
{
  __shared__ __align__(16) __bf16 As[16 * 512];
  __shared__ int rowdst[16];
  const int Kpad = KT * 32;
  const int tile = blockIdx.x;
  const int tid  = threadIdx.x;

  if (tid < 16) {
    int r = tile * 16 + tid;
    int rd = -1;
    if (r < nrows) {
      if (MODE == 0)      rd = jidx[r];
      else if (MODE == 1) rd = dstp[r];
      else                rd = r;
    }
    rowdst[tid] = rd;
  }

  // Cooperative gather/concat of the 16 A rows into LDS (bf16, K zero-padded).
  for (int idx = tid; idx < 16 * Kpad; idx += 256) {
    int row = idx / Kpad;
    int k   = idx - row * Kpad;
    int r   = tile * 16 + row;
    float v = 0.0f;
    if (r < nrows) {
      if (MODE == 0) {
        if (k < in_c)            v = hsrc[(long)kidx[r] * hld + k];
        else if (k < in_c + 6)   v = rbf[(long)jidx[r] * 6 + (k - in_c)];
        else if (k < in_c + 12)  v = cbf[(long)r * 6 + (k - in_c - 6)];
      } else if (MODE == 1) {
        if (k < in_c) v = hsrc[(long)srcp[r] * hld + k];
        else          v = aux[(long)dstp[r] * 256 + (k - in_c)];
      } else {
        if (k < in_c) v = hsrc[(long)r * hld + k];
        else          v = aux[(long)r * 256 + (k - in_c)];
      }
    }
    As[row * Kpad + k] = (__bf16)v;
  }
  __syncthreads();

  const int wave = tid >> 5;     // wave32
  const int lane = tid & 31;
  const int nt0  = wave * 2;
  const int arow = lane & 15;
  const int kofs = (lane < 16) ? 0 : 8;   // ISA 16-bit A layout

  v8f acc0 = {0.f, 0.f, 0.f, 0.f, 0.f, 0.f, 0.f, 0.f};
  v8f acc1 = {0.f, 0.f, 0.f, 0.f, 0.f, 0.f, 0.f, 0.f};

  for (int kt = 0; kt < KT; ++kt) {
    // A fragment: halves 0..7 = K kt*32+kofs .. +7 ; 8..15 = +16 .. +23
    const __bf16* ap = &As[arow * Kpad + kt * 32 + kofs];
    union { v16bf v; v8bf h[2]; } au;
    au.h[0] = *(const v8bf*)ap;          // 16B aligned ds_load_b128
    au.h[1] = *(const v8bf*)(ap + 16);
    // B fragments: 32B contiguous per lane (fragment-major packed weights)
    long boff = (((long)(kt * 16 + nt0) * 32) + lane) * 16;
    v16bf b0 = *(const v16bf*)(wp + boff);
    v16bf b1 = *(const v16bf*)(wp + boff + 512);
    acc0 = __builtin_amdgcn_wmma_f32_16x16x32_bf16(false, au.v, false, b0,
                                                   (short)0, acc0, false, false);
    acc1 = __builtin_amdgcn_wmma_f32_16x16x32_bf16(false, au.v, false, b1,
                                                   (short)0, acc1, false, false);
  }

  // C/D layout: lane<16 -> (M=v,   N=nt*16+lane); lane>=16 -> (M=v+8, N=nt*16+lane-16)
  const int colb = nt0 * 16 + (lane & 15);
  const int rb   = (lane < 16) ? 0 : 8;

  if (MODE == 2) {
    float bb0 = bias[colb];
    float bb1 = bias[colb + 16];
#pragma unroll
    for (int v = 0; v < 8; ++v) {
      int rd = rowdst[rb + v];
      if (rd >= 0) {
        outbuf[(long)rd * 256 + colb]      = acc0[v] + bb0;
        outbuf[(long)rd * 256 + colb + 16] = acc1[v] + bb1;
      }
    }
  } else {
#pragma unroll
    for (int v = 0; v < 8; ++v) {
      int rd = rowdst[rb + v];
      if (rd >= 0) {
        float x0 = acc0[v]; x0 = x0 > 0.f ? x0 : 0.f;
        float x1 = acc1[v]; x1 = x1 > 0.f ? x1 : 0.f;
        atomic_add_f32(&outbuf[(long)rd * 256 + colb],      x0);
        atomic_add_f32(&outbuf[(long)rd * 256 + colb + 16], x1);
      }
    }
  }
}

// ---------------------------------------------------------------------------
// Per-graph mean-pool accumulation (sums + counts via fp32 atomics).
// ---------------------------------------------------------------------------
__global__ void pool_kernel(const float* __restrict__ h, const int* __restrict__ batch,
                            float* __restrict__ pooled, float* __restrict__ cnt, int N) {
  int i = blockIdx.x;
  int c = threadIdx.x;
  if (i < N) {
    int g = batch[i];
    atomic_add_f32(&pooled[(long)g * 256 + c], h[(long)i * 256 + c]);
    if (c == 0) atomic_add_f32(&cnt[g], 1.0f);
  }
}

// ---------------------------------------------------------------------------
// Head: mean, ReLU, ReLU(g@W1+b1)@W2+b2  -> out [G,128]. Tiny (G=64), plain VALU.
// ---------------------------------------------------------------------------
__global__ __launch_bounds__(256)
void head_kernel(const float* __restrict__ pooled, const float* __restrict__ cnt,
                 const float* __restrict__ W1, const float* __restrict__ b1,
                 const float* __restrict__ W2, const float* __restrict__ b2,
                 float* __restrict__ out) {
  __shared__ float g[256];
  __shared__ float t1[256];
  int gid = blockIdx.x;
  int tid = threadIdx.x;
  float c = cnt[gid];
  c = (c > 1.0f) ? c : 1.0f;
  float v = pooled[(long)gid * 256 + tid] / c;
  g[tid] = (v > 0.f) ? v : 0.f;
  __syncthreads();
  float acc = b1[tid];
  for (int k = 0; k < 256; ++k) acc += g[k] * W1[k * 256 + tid];
  t1[tid] = (acc > 0.f) ? acc : 0.f;
  __syncthreads();
  if (tid < 128) {
    float a2 = b2[tid];
    for (int k = 0; k < 256; ++k) a2 += t1[k] * W2[k * 128 + tid];
    out[(long)gid * 128 + tid] = a2;
  }
}

// ---------------------------------------------------------------------------
extern "C" void kernel_launch(void* const* d_in, const int* in_sizes, int n_in,
                              void* d_out, int out_size, void* d_ws, size_t ws_size,
                              hipStream_t stream) {
  (void)in_sizes; (void)n_in; (void)out_size; (void)ws_size;
  constexpr int N = 20000, E = 250000, T = 250000, G = 64;

  const float* x     = (const float*)d_in[0];
  const int*   ei    = (const int*)d_in[1];
  const float* rbf   = (const float*)d_in[2];
  const float* cbf   = (const float*)d_in[3];
  const int*   kidx  = (const int*)d_in[4];
  const int*   jidx  = (const int*)d_in[5];
  /* d_in[6] = i_idx: unused by reference */
  const int*   batch = (const int*)d_in[7];

  const int* srcp = ei;
  const int* dstp = ei + E;

  // ---- weight pack metadata (all GEMMs have 256 output cols) ----
  const int KT_e1[3] = {5, 9, 9};       // K=140 -> 160, K=268 -> 288
  const int K_e1 [3] = {140, 268, 268};
  const int KT_m [3] = {12, 16, 16};    // K=384, K=512 (already x32)
  const int K_m  [3] = {384, 512, 512};

  size_t off = 0, off_e1[3], off_e2[3], off_n[3];
  for (int l = 0; l < 3; ++l) {
    off_e1[l] = off; off += (size_t)KT_e1[l] * 8192;   // KT * 16 tiles * 512 halves
    off_e2[l] = off; off += (size_t)KT_m [l] * 8192;
    off_n [l] = off; off += (size_t)KT_m [l] * 8192;
  }

  // ---- workspace layout: packed bf16 weights first, then fp32 buffers ----
  __bf16* wpack = (__bf16*)d_ws;
  size_t wp_bytes = ((off * 2 + 255) / 256) * 256;
  float* fbase     = (float*)((char*)d_ws + wp_bytes);
  float* h0        = fbase;
  float* h1        = h0 + (size_t)N * 256;
  float* agg_e     = h1 + (size_t)N * 256;           // [E,256]
  float* node_aggr = agg_e + (size_t)E * 256;        // [N,256]
  float* pooled    = node_aggr + (size_t)N * 256;    // [G,256]
  float* cnt       = pooled + (size_t)G * 256;       // [G]

  // ---- pack all weights to fragment-major bf16 (deterministic each call) ----
  for (int l = 0; l < 3; ++l) {
    const float* We1 = (const float*)d_in[8 + 4 * l];
    const float* We2 = (const float*)d_in[9 + 4 * l];
    const float* Wn  = (const float*)d_in[10 + 4 * l];
    int t1n = KT_e1[l] * 8192;
    int t2n = KT_m[l] * 8192;
    pack_weight_bf16<<<(t1n + 255) / 256, 256, 0, stream>>>(We1, wpack + off_e1[l], K_e1[l], KT_e1[l]);
    pack_weight_bf16<<<(t2n + 255) / 256, 256, 0, stream>>>(We2, wpack + off_e2[l], K_m[l], KT_m[l]);
    pack_weight_bf16<<<(t2n + 255) / 256, 256, 0, stream>>>(Wn,  wpack + off_n[l],  K_m[l], KT_m[l]);
  }

  // ---- 3 message-passing layers ----
  const float* hsrc = x;
  int hld = 128;
  float* hbufs[2] = {h0, h1};
  for (int l = 0; l < 3; ++l) {
    int in_c = (l == 0) ? 128 : 256;
    float* hdst = hbufs[l & 1];
    const float* bn = (const float*)d_in[11 + 4 * l];

    hipMemsetAsync(agg_e, 0, (size_t)E * 256 * sizeof(float), stream);
    fused_gather_gemm_scatter<0><<<(T + 15) / 16, 256, 0, stream>>>(
        hsrc, hld, in_c, rbf, cbf, kidx, jidx, srcp, dstp,
        nullptr, wpack + off_e1[l], KT_e1[l], nullptr, agg_e, T);

    hipMemsetAsync(node_aggr, 0, (size_t)N * 256 * sizeof(float), stream);
    fused_gather_gemm_scatter<1><<<(E + 15) / 16, 256, 0, stream>>>(
        hsrc, hld, in_c, rbf, cbf, kidx, jidx, srcp, dstp,
        agg_e, wpack + off_e2[l], KT_m[l], nullptr, node_aggr, E);

    fused_gather_gemm_scatter<2><<<(N + 15) / 16, 256, 0, stream>>>(
        hsrc, hld, in_c, rbf, cbf, kidx, jidx, srcp, dstp,
        node_aggr, wpack + off_n[l], KT_m[l], bn, hdst, N);

    hsrc = hdst;
    hld = 256;
  }

  // ---- mean pool + head MLP ----
  hipMemsetAsync(pooled, 0, (size_t)(G * 256 + G) * sizeof(float), stream);
  pool_kernel<<<N, 256, 0, stream>>>(hsrc, batch, pooled, cnt, N);
  head_kernel<<<G, 256, 0, stream>>>(pooled, cnt,
                                     (const float*)d_in[20], (const float*)d_in[21],
                                     (const float*)d_in[22], (const float*)d_in[23],
                                     (float*)d_out);
}
